// VQVAE_61830349193407
// MI455X (gfx1250) — matrix operand order repaired and checked by reading, not compiled, not run on hardware
//
#include <hip/hip_runtime.h>
#include <hip/hip_bf16.h>

typedef __attribute__((ext_vector_type(16))) _Float16 v16h;
typedef __attribute__((ext_vector_type(8)))  _Float16 v8h;
typedef __attribute__((ext_vector_type(8)))  float    v8f;

#define BM 128
#define BN 64
#define BK 32
#define LDSS 40   // LDS row stride in halves (32 + 8 pad; 40*2=80=16*5 keeps b128 alignment)

#if defined(__gfx1250__) && __has_builtin(__builtin_amdgcn_global_load_async_to_lds_b128) && \
    __has_builtin(__builtin_amdgcn_s_wait_asynccnt)
#define USE_ASYNC_LDS 1
#else
#define USE_ASYNC_LDS 0
#endif

#if USE_ASYNC_LDS
typedef int v4i_vs __attribute__((vector_size(16)));
typedef __attribute__((address_space(1))) v4i_vs* g_v4i_ptr;
typedef __attribute__((address_space(3))) v4i_vs* l_v4i_ptr;
#endif

__device__ __forceinline__ void async_cp16(const _Float16* g, _Float16* l) {
#if USE_ASYNC_LDS
  __builtin_amdgcn_global_load_async_to_lds_b128(
      (g_v4i_ptr)(unsigned long long)(const void*)g,
      (l_v4i_ptr)(unsigned)(unsigned long long)(const void*)l,
      0, 0);
#else
  (void)g; (void)l;
#endif
}

// ---------------- weight convert: f32 [K][N] -> f16 transposed padded [Np64][Kp] ---------
__global__ void cvt_wt(const float* __restrict__ W, _Float16* __restrict__ out,
                       int K, int N, int Kp, int Np64) {
  int idx = blockIdx.x * blockDim.x + threadIdx.x;
  if (idx >= Kp * Np64) return;
  int n = idx / Kp, k = idx - n * Kp;
  float v = (k < K && n < N) ? W[k * N + n] : 0.0f;
  out[idx] = (_Float16)v;
}

// ---------------- x convert: f32 [B][K] -> f16 [B][Kp] ----------------
__global__ void cvt_x(const float* __restrict__ x, _Float16* __restrict__ out,
                      int K, int Kp, long total) {
  long idx = (long)blockIdx.x * blockDim.x + threadIdx.x;
  if (idx >= total) return;
  long b = idx / Kp;
  int  k = (int)(idx - b * Kp);
  out[idx] = (k < K) ? (_Float16)x[b * K + k] : (_Float16)0.0f;
}

// ---------------- WMMA GEMM:  out = act(A[B][Kp] @ Wt^T + bias) -------------------------
// Wt stored transposed + zero-padded: [>= gridDim.x*BN rows][Kp cols], f16.
// MODE 0: ReLU -> f16 out (ld=ldo, zero-fill cols [Ntrue,ldo))
// MODE 1: none -> f32 out (cols < Ntrue only)
// MODE 2: sigmoid -> f32 out (cols < Ntrue only)
template <int MODE>
__global__ __launch_bounds__(256)
void gemm_wmma(const _Float16* __restrict__ A, int Kp,
               const _Float16* __restrict__ Wt,
               const float* __restrict__ bias, int Ntrue,
               void* __restrict__ outp, int ldo) {
  __shared__ __align__(16) _Float16 sA[2][BM * LDSS];
  __shared__ __align__(16) _Float16 sB[2][BN * LDSS];

  const int tid  = threadIdx.x;
  const int lane = tid & 31;
  const int wid  = tid >> 5;     // 8 waves
  const int wm   = wid >> 1;     // 0..3  (M sub-tile, 32 rows each)
  const int wn   = wid & 1;      // 0..1  (N sub-tile, 32 cols each)
  const int g    = lane >> 4;    // lane half-group
  const int lr   = lane & 15;

  const long rowBase = (long)blockIdx.y * BM;
  const int  colBase = blockIdx.x * BN;

  // copy coordinates: 16-byte (8-half) chunks
  // A tile: 128 rows x 4 chunks = 512 chunks, 2 per thread
  // B tile:  64 rows x 4 chunks = 256 chunks, 1 per thread
  const int rB  = tid >> 2;
  const int cB  = (tid & 3) * 8;

#if !USE_ASYNC_LDS
  int4 regA[2];
  int4 regB;
#endif

  auto stage = [&](int k0, int buf) {
#pragma unroll
    for (int i = 0; i < 2; ++i) {
      int c = tid + i * 256;
      int r = c >> 2;
      int c8 = (c & 3) * 8;
#if USE_ASYNC_LDS
      async_cp16(A + (rowBase + r) * Kp + k0 + c8, &sA[buf][r * LDSS + c8]);
#else
      regA[i] = *(const int4*)(A + (rowBase + r) * Kp + k0 + c8);
#endif
    }
#if USE_ASYNC_LDS
    async_cp16(Wt + (long)(colBase + rB) * Kp + k0 + cB, &sB[buf][rB * LDSS + cB]);
#else
    regB = *(const int4*)(Wt + (long)(colBase + rB) * Kp + k0 + cB);
#endif
  };

  auto commit = [&](int buf) {
#if USE_ASYNC_LDS
    __builtin_amdgcn_s_wait_asynccnt(0);
    (void)buf;
#else
#pragma unroll
    for (int i = 0; i < 2; ++i) {
      int c = tid + i * 256;
      int r = c >> 2;
      int c8 = (c & 3) * 8;
      *(int4*)(&sA[buf][r * LDSS + c8]) = regA[i];
    }
    *(int4*)(&sB[buf][rB * LDSS + cB]) = regB;
#endif
  };

  v8f acc[2][2] = {};

  stage(0, 0);
  commit(0);
  __syncthreads();

  int buf = 0;
  for (int k0 = 0; k0 < Kp; k0 += BK) {
    const int kn = k0 + BK;
    if (kn < Kp) stage(kn, buf ^ 1);

    // fragments per CDNA5 WMMA VGPR layout
    v16h af[2], bf[2];
#pragma unroll
    for (int r = 0; r < 2; ++r) {
      const _Float16* p = &sA[buf][(wm * 32 + r * 16 + lr) * LDSS];
      v8h lo = *(const v8h*)(p + g * 8);        // K = g*8 .. g*8+7
      v8h hi = *(const v8h*)(p + 16 + g * 8);   // K = 16+g*8 ..
      af[r] = __builtin_shufflevector(lo, hi, 0, 1, 2, 3, 4, 5, 6, 7,
                                      8, 9, 10, 11, 12, 13, 14, 15);
    }
#pragma unroll
    for (int c = 0; c < 2; ++c) {
      const _Float16* p = &sB[buf][(wn * 32 + c * 16 + lr) * LDSS + g * 8];
      v8h lo = *(const v8h*)(p);                // K = g*8 .. +7
      v8h hi = *(const v8h*)(p + 16);           // K = 16+g*8 .. +7
      bf[c] = __builtin_shufflevector(lo, hi, 0, 1, 2, 3, 4, 5, 6, 7,
                                      8, 9, 10, 11, 12, 13, 14, 15);
    }
#pragma unroll
    for (int r = 0; r < 2; ++r)
#pragma unroll
      for (int c = 0; c < 2; ++c)
        acc[r][c] = __builtin_amdgcn_wmma_f32_16x16x32_f16(
            false, af[r], false, bf[c], (short)0, acc[r][c], false, false);

    if (kn < Kp) {
      commit(buf ^ 1);
      __syncthreads();
      buf ^= 1;
    }
  }

  // epilogue (C/D layout: N = lane&15, M = g*8 + vi)
#pragma unroll
  for (int r = 0; r < 2; ++r) {
#pragma unroll
    for (int c = 0; c < 2; ++c) {
      int n = colBase + wn * 32 + c * 16 + lr;
      float bn = (n < Ntrue) ? bias[n] : 0.0f;
#pragma unroll
      for (int vi = 0; vi < 8; ++vi) {
        long m = rowBase + wm * 32 + r * 16 + g * 8 + vi;
        float val = acc[r][c][vi] + bn;
        if (MODE == 0) {
          if (n < ldo) {
            _Float16 o = (n < Ntrue) ? (_Float16)fmaxf(val, 0.0f)
                                     : (_Float16)0.0f;
            ((_Float16*)outp)[m * ldo + n] = o;
          }
        } else {
          if (n < Ntrue) {
            float o = (MODE == 2) ? (1.0f / (1.0f + __expf(-val))) : val;
            ((float*)outp)[m * (long)ldo + n] = o;
          }
        }
      }
    }
  }
}

// ---------------- VQ nearest-embedding: z (B,10,20) vs codebook (10,10) -----------------
__global__ void vq_kernel(const float* __restrict__ z, const float* __restrict__ cb,
                          float* __restrict__ emb, _Float16* __restrict__ zqh,
                          long total) {
  __shared__ float w[100];
  int tid = threadIdx.x;
  if (tid < 100) w[tid] = cb[tid];
  __syncthreads();
  long idx = (long)blockIdx.x * blockDim.x + tid;
  if (idx >= total) return;
  long b = idx / 20;
  int  d = (int)(idx - b * 20);
  const float* zb = z + b * 200 + d;
  float v[10];
#pragma unroll
  for (int e = 0; e < 10; ++e) v[e] = zb[e * 20];
  int best = 0;
  float bestd = 3.4e38f;
#pragma unroll
  for (int k = 0; k < 10; ++k) {
    float s = 0.f;
#pragma unroll
    for (int e = 0; e < 10; ++e) {
      float df = v[e] - w[e * 10 + k];
      s += df * df;
    }
    if (s < bestd) { bestd = s; best = k; }
  }
  float*    eb = emb + b * 200 + d;
  _Float16* qb = zqh + b * 224 + d;
#pragma unroll
  for (int e = 0; e < 10; ++e) {
    float val = w[e * 10 + best];
    eb[e * 20] = val;
    qb[e * 20] = (_Float16)val;
  }
}

__global__ void pad_zq(_Float16* __restrict__ zqh, long total) {
  long idx = (long)blockIdx.x * blockDim.x + threadIdx.x;
  if (idx >= total) return;
  long b = idx / 24;
  int  c = 200 + (int)(idx - b * 24);
  zqh[b * 224 + c] = (_Float16)0.0f;
}

// ----------------------------------- host launcher --------------------------------------
extern "C" void kernel_launch(void* const* d_in, const int* in_sizes, int n_in,
                              void* d_out, int out_size, void* d_ws, size_t ws_size,
                              hipStream_t stream) {
  const float* x    = (const float*)d_in[0];
  const float* We1  = (const float*)d_in[1];  const float* be1 = (const float*)d_in[2];
  const float* We2  = (const float*)d_in[3];  const float* be2 = (const float*)d_in[4];
  const float* We3  = (const float*)d_in[5];  const float* be3 = (const float*)d_in[6];
  const float* Wd1  = (const float*)d_in[7];  const float* bd1 = (const float*)d_in[8];
  const float* Wd2  = (const float*)d_in[9];  const float* bd2 = (const float*)d_in[10];
  const float* Wd3  = (const float*)d_in[11]; const float* bd3 = (const float*)d_in[12];
  const float* Wd4  = (const float*)d_in[13]; const float* bd4 = (const float*)d_in[14];
  const float* embw = (const float*)d_in[15];

  const long B = in_sizes[0] / 784;

  size_t cur = 0;
  auto alloc = [&](size_t bytes) -> void* {
    cur = (cur + 255) & ~(size_t)255;
    void* p = (char*)d_ws + cur;
    cur += bytes;
    return p;
  };

  _Float16* xh = (_Float16*)alloc((size_t)B * 800 * 2);
  _Float16* h1 = (_Float16*)alloc((size_t)B * 512 * 2);
  _Float16* h2 = (_Float16*)alloc((size_t)B * 320 * 2);
  _Float16* zq = (_Float16*)alloc((size_t)B * 224 * 2);
  // weights transposed, rows padded to multiple of 64 (covers all N-tiles: no load guards)
  _Float16* we1t = (_Float16*)alloc((size_t)512 * 800 * 2);
  _Float16* we2t = (_Float16*)alloc((size_t)320 * 512 * 2);
  _Float16* we3t = (_Float16*)alloc((size_t)256 * 320 * 2);
  _Float16* wd1t = (_Float16*)alloc((size_t)256 * 224 * 2);
  _Float16* wd2t = (_Float16*)alloc((size_t)320 * 224 * 2);
  _Float16* wd3t = (_Float16*)alloc((size_t)512 * 320 * 2);
  _Float16* wd4t = (_Float16*)alloc((size_t)832 * 512 * 2);
  _Float16* dd1 = h2;  // reuse (h2 dead after z)
  _Float16* dd2 = h1;  // reuse (h1 dead after layer 2)
  _Float16* dd3 = xh;  // reuse (xh dead after layer 1)

  float* zout   = (float*)d_out + (size_t)B * 784;  // z_e slot
  float* embout = zout + (size_t)B * 200;           // emb slot

  // ---- weight conversion (transpose + pad) ----
  auto cvtw = [&](const float* W, _Float16* o, int K, int N, int Kp, int Np64) {
    int tot = Kp * Np64;
    hipLaunchKernelGGL(cvt_wt, dim3((tot + 255) / 256), dim3(256), 0, stream,
                       W, o, K, N, Kp, Np64);
  };
  cvtw(We1, we1t, 784, 500, 800, 512);
  cvtw(We2, we2t, 500, 300, 512, 320);
  cvtw(We3, we3t, 300, 200, 320, 256);
  cvtw(Wd1, wd1t, 200, 200, 224, 256);
  cvtw(Wd2, wd2t, 200, 300, 224, 320);
  cvtw(Wd3, wd3t, 300, 500, 320, 512);
  cvtw(Wd4, wd4t, 500, 784, 512, 832);

  // ---- x -> f16 padded ----
  {
    long tot = B * 800;
    hipLaunchKernelGGL(cvt_x, dim3((unsigned)((tot + 255) / 256)), dim3(256), 0, stream,
                       x, xh, 784, 800, tot);
  }

  const dim3 blk(256);
  const unsigned gy = (unsigned)(B / BM);

  // encoder
  hipLaunchKernelGGL((gemm_wmma<0>), dim3(8, gy),  blk, 0, stream,
                     xh, 800, we1t, be1, 500, (void*)h1, 512);
  hipLaunchKernelGGL((gemm_wmma<0>), dim3(5, gy),  blk, 0, stream,
                     h1, 512, we2t, be2, 300, (void*)h2, 320);
  hipLaunchKernelGGL((gemm_wmma<1>), dim3(4, gy),  blk, 0, stream,
                     h2, 320, we3t, be3, 200, (void*)zout, 200);

  // VQ lookup (z_e already in d_out; write emb + f16 z_q)
  {
    long tot = B * 20;
    hipLaunchKernelGGL(vq_kernel, dim3((unsigned)((tot + 255) / 256)), blk, 0, stream,
                       zout, embw, embout, zq, tot);
    long ptot = B * 24;
    hipLaunchKernelGGL(pad_zq, dim3((unsigned)((ptot + 255) / 256)), blk, 0, stream,
                       zq, ptot);
  }

  // decoder
  hipLaunchKernelGGL((gemm_wmma<0>), dim3(4, gy),  blk, 0, stream,
                     zq, 224, wd1t, bd1, 200, (void*)dd1, 224);
  hipLaunchKernelGGL((gemm_wmma<0>), dim3(5, gy),  blk, 0, stream,
                     dd1, 224, wd2t, bd2, 300, (void*)dd2, 320);
  hipLaunchKernelGGL((gemm_wmma<0>), dim3(8, gy),  blk, 0, stream,
                     dd2, 320, wd3t, bd3, 500, (void*)dd3, 512);
  hipLaunchKernelGGL((gemm_wmma<2>), dim3(13, gy), blk, 0, stream,
                     dd3, 512, wd4t, bd4, 784, (void*)d_out, 784);

  (void)n_in; (void)out_size; (void)ws_size;
}
